// LatticeLSTM_58583353917659
// MI455X (gfx1250) — matrix operand order repaired and checked by reading, not compiled, not run on hardware
//
#include <hip/hip_runtime.h>
#include <hip/hip_bf16.h>

#define B_  32
#define S_  1024
#define W_  256
#define E_  256
#define H_  512
#define H2_ 256
#define L_  32

typedef __attribute__((ext_vector_type(16))) __bf16         v16bf;
typedef __attribute__((ext_vector_type(8)))  float          v8f;
typedef __attribute__((ext_vector_type(8)))  unsigned short u16x8;
typedef __attribute__((ext_vector_type(4)))  unsigned int   v4u;
typedef __attribute__((ext_vector_type(8)))  int            v8i;
typedef __attribute__((ext_vector_type(4)))  int            v4i;

union Frag { v16bf bf; u16x8 h[2]; };
union AccU { v8f v; float f[8]; };

__device__ __forceinline__ unsigned short f2bf(float f) {
  unsigned int u = __builtin_bit_cast(unsigned int, f);
  u += 0x7FFFu + ((u >> 16) & 1u);            // round-to-nearest-even
  return (unsigned short)(u >> 16);
}
__device__ __forceinline__ float sigf(float x) { return 1.0f / (1.0f + __expf(-x)); }
__device__ __forceinline__ float tanhfast(float x) {
  x = fminf(fmaxf(x, -15.0f), 15.0f);
  float e = __expf(2.0f * x);
  return (e - 1.0f) / (e + 1.0f);
}

// A-fragment (16x32 bf16, M x K) from LDS. Lane l: row = l&15,
// halves 0..7  -> K = kb + (l>>4)*8 + 0..7
// halves 8..15 -> K = kb + 16 + (l>>4)*8 + 0..7
__device__ __forceinline__ Frag load_a_lds(const unsigned short* lds, int stride, int kb) {
  const int l   = threadIdx.x & 31;
  const int row = l & 15;
  const int g   = l >> 4;
  const unsigned short* p = lds + row * stride + kb + g * 8;
  Frag f;
  f.h[0] = *(const u16x8*)p;
  f.h[1] = *(const u16x8*)(p + 16);
  return f;
}

// B-fragment (32x16 bf16, K x N) from pre-packed global tiles:
// tile = 32 lanes * 16 contiguous halves (lane l holds col l&15, K = (l>>4)*16 + h)
__device__ __forceinline__ Frag load_b(const unsigned short* Bp, int tile) {
  const int l = threadIdx.x & 31;
  const unsigned short* p = Bp + (size_t)tile * 512 + (l & 15) * 16 + (l >> 4) * 8;
  Frag f;
  f.h[0] = *(const u16x8*)p;
  f.h[1] = *(const u16x8*)(p + 256);
  return f;
}

__device__ __forceinline__ v8f wmma_bf16(const Frag& a, const Frag& b, v8f c) {
  return __builtin_amdgcn_wmma_f32_16x16x32_bf16(false, a.bf, false, b.bf,
                                                 (short)0, c, false, false);
}

// ---------------- embedding gather -> bf16 (E_ = 256 cols) ----------------
__global__ void k_embed(const int* __restrict__ ids, const float* __restrict__ emb,
                        unsigned short* __restrict__ X, int rows) {
  int gid = blockIdx.x * blockDim.x + threadIdx.x;
  if (gid >= rows * E_) return;
  int row = gid >> 8, col = gid & 255;
  X[gid] = f2bf(emb[(size_t)ids[row] * E_ + col]);
}

// -------- pack W[N,K] (f32, row n = output, col k = reduce) into B tiles ----
__global__ void k_pack(const float* __restrict__ Wm, unsigned short* __restrict__ P,
                       int N, int K) {
  int gid = blockIdx.x * blockDim.x + threadIdx.x;
  if (gid >= N * K) return;
  int h    = gid & 15;
  int lane = (gid >> 4) & 31;
  int tile = gid >> 9;
  int NT = N >> 4;
  int nt = tile % NT, kt = tile / NT;
  int k = (kt << 5) + ((lane >> 4) << 4) + h;
  int n = (nt << 4) + (lane & 15);
  P[gid] = f2bf(Wm[(size_t)n * K + k]);
}

// ------------- generic WMMA GEMM: C[M,N] = A(bf16)[M,K] * Bpacked + bias ----
// act: 0 = none, 1 = sigmoid
// A tile (16 rows x K halves) staged into LDS by the Tensor Data Mover with
// hardware row padding: 4 DWORDs inserted every min(K,512) halves.
__global__ void k_gemm(const unsigned short* __restrict__ A,
                       const unsigned short* __restrict__ Bp,
                       float* __restrict__ C,
                       const float* __restrict__ bias1, const float* __restrict__ bias2,
                       int M, int N, int K, int act) {
  __shared__ __align__(16) unsigned short As[16 * (1024 + 16) + 16];
  const int m0 = blockIdx.x * 16;
  const int PADI      = (K <= 512) ? K : 512;      // halves between pads
  const int rowStride = K + 8 * (K / PADI);        // halves per row in LDS

#if defined(__has_builtin) && __has_builtin(__builtin_amdgcn_tensor_load_to_lds) && \
    __has_builtin(__builtin_amdgcn_s_wait_tensorcnt)
  if (threadIdx.x < 32) {
    unsigned long long ga = (unsigned long long)(uintptr_t)(A + (size_t)m0 * K);
    unsigned lds_off = (unsigned)(uintptr_t)(void*)As;   // low 32 bits = LDS byte offset
    v4u g0;
    g0[0] = 1u;                                          // count=1 (user descriptor)
    g0[1] = lds_off;                                     // lds_addr
    g0[2] = (unsigned)(ga & 0xFFFFFFFFu);                // global_addr[31:0]
    g0[3] = (unsigned)((ga >> 32) & 0x01FFFFFFu) | (2u << 30);  // addr[56:32], type=2
    const int padi_code = (K <= 256) ? 6 : 7;            // 128 / 256 DWORD interval
    v8i g1;
    g1[0] = (1 << 16)            // data_size = 2 bytes
          | (1 << 20)            // pad_enable
          | (padi_code << 22)    // pad_interval
          | (3 << 25);           // pad_amount = 4 DWORDs (8 halves)
    g1[1] = (K & 0xFFFF) << 16;                          // tensor_dim0[15:0]
    g1[2] = ((K >> 16) & 0xFFFF) | (16 << 16);           // dim0 hi | tensor_dim1=16
    g1[3] = (K & 0xFFFF) << 16;                          // dim1 hi=0 | tile_dim0=K
    g1[4] = 16;                                          // tile_dim1=16, tile_dim2=0
    g1[5] = K;                                           // tensor_dim0_stride[31:0]
    g1[6] = 0;                                           // stride hi | dim1_stride lo
    g1[7] = 0;
    v4i g2 = {0, 0, 0, 0};
    v4i g3 = {0, 0, 0, 0};
    v8i g4 = {0, 0, 0, 0, 0, 0, 0, 0};                   // 6-arg toolchain variant
    __builtin_amdgcn_tensor_load_to_lds(g0, g1, g2, g3, g4, 0);
    __builtin_amdgcn_s_wait_tensorcnt(0);
  }
#else
  {
    const int K2 = K >> 1, P2 = rowStride >> 1;
    const unsigned int* src = (const unsigned int*)(A + (size_t)m0 * K);
    unsigned int* dst = (unsigned int*)As;
    for (int i = threadIdx.x; i < 16 * K2; i += blockDim.x) {
      int r = i / K2, c = i - r * K2;
      int cp = c + ((K > 512 && c >= 256) ? 4 : 0);      // mid-row pad for K=1024
      dst[r * P2 + cp] = src[i];
    }
  }
#endif
  __syncthreads();

  const int wave = threadIdx.x >> 5;
  const int lane = threadIdx.x & 31;
  const int wpb  = blockDim.x >> 5;
  const int NT   = N >> 4;
  const int nt   = blockIdx.y * wpb + wave;
  if (nt >= NT) return;

  const int KT = K >> 5;
  v8f acc = {0.f,0.f,0.f,0.f,0.f,0.f,0.f,0.f};
  for (int kc = 0; kc < KT; ++kc) {
    const int kb  = kc << 5;
    const int kbp = kb + ((kb & 512) ? 8 : 0);           // skip mid-row pad (K=1024)
    if (kc + 1 < KT)
      __builtin_prefetch(Bp + ((size_t)((kc + 1) * NT + nt) * 512) + (lane << 4), 0, 1);
    Frag a = load_a_lds(As, rowStride, kbp);
    Frag b = load_b(Bp, kc * NT + nt);
    acc = wmma_bf16(a, b, acc);
  }

  const int n = (nt << 4) + (lane & 15);
  float bsum = bias1 ? bias1[n] : 0.f;
  if (bias2) bsum += bias2[n];
  AccU u; u.v = acc;
  const int mh = (lane >> 4) << 3;
#pragma unroll
  for (int r = 0; r < 8; ++r) {
    float v = u.f[r] + bsum;
    if (act == 1) v = sigf(v);
    C[(size_t)(m0 + mh + r) * N + n] = v;
  }
}

// ------------- persistent recurrent LSTM over one direction ----------------
// xp:  [B, T, 1024] f32 precomputed x@Wih^T + bih + bhh
// out: [B, T, 512]  f32, columns [colOff, colOff+256)
// Block = 16 batch rows, 16 waves; wave w owns hidden cols [16w,16w+16) of all
// four gates; c lives in registers, h round-trips through LDS as bf16.
__global__ void __launch_bounds__(512, 1)
k_lstm(const float* __restrict__ xp, const unsigned short* __restrict__ WhhP,
       float* __restrict__ out, int T, int rev, int colOff) {
  __shared__ __align__(16) unsigned short hsm[16 * 264];
  const int wave = threadIdx.x >> 5;
  const int lane = threadIdx.x & 31;
  const int m0   = blockIdx.x * 16;
  for (int i = threadIdx.x; i < 16 * 264; i += blockDim.x) hsm[i] = 0;
  AccU c_reg;
#pragma unroll
  for (int r = 0; r < 8; ++r) c_reg.f[r] = 0.f;
  __syncthreads();

  const int mloc = (lane >> 4) << 3;                // + r gives local row
  const int ncol = (wave << 4) + (lane & 15);       // hidden column 0..255

  for (int t = 0; t < T; ++t) {
    const int tt = rev ? (T - 1 - t) : t;
    // latency hint for next step's gate pre-activations
    if (t + 1 < T) {
      const int tn = rev ? (T - 2 - t) : (t + 1);
      __builtin_prefetch(&xp[((size_t)(m0 + mloc) * T + tn) * 1024 + ncol], 0, 1);
    }
    Frag a[8];
#pragma unroll
    for (int kc = 0; kc < 8; ++kc) a[kc] = load_a_lds(hsm, 264, kc << 5);

    v8f acc[4];
#pragma unroll
    for (int gi = 0; gi < 4; ++gi) {
      v8f z = {0.f,0.f,0.f,0.f,0.f,0.f,0.f,0.f};
#pragma unroll
      for (int kc = 0; kc < 8; ++kc) {
        Frag b = load_b(WhhP, kc * 64 + gi * 16 + wave);
        z = wmma_bf16(a[kc], b, z);
      }
      acc[gi] = z;
    }
    AccU ai, af, ag, ao;
    ai.v = acc[0]; af.v = acc[1]; ag.v = acc[2]; ao.v = acc[3];

    float hv[8];
#pragma unroll
    for (int r = 0; r < 8; ++r) {
      const int m = m0 + mloc + r;
      const size_t base = ((size_t)m * T + tt) * 1024;
      float zi = ai.f[r] + xp[base + ncol];
      float zf = af.f[r] + xp[base + 256 + ncol];
      float zg = ag.f[r] + xp[base + 512 + ncol];
      float zo = ao.f[r] + xp[base + 768 + ncol];
      float c = sigf(zf) * c_reg.f[r] + sigf(zi) * tanhfast(zg);
      float h = sigf(zo) * tanhfast(c);
      c_reg.f[r] = c;
      hv[r] = h;
      out[((size_t)m * T + tt) * 512 + colOff + ncol] = h;
    }
    __syncthreads();   // all A-fragment reads done before overwrite
#pragma unroll
    for (int r = 0; r < 8; ++r)
      hsm[(mloc + r) * 264 + ncol] = f2bf(hv[r]);
    __syncthreads();   // h visible to all waves for next step
  }
}

// ------------- span mean + concat -> gate-GEMM A matrix (bf16) -------------
__global__ void k_span(const int* __restrict__ wpos, const float* __restrict__ cout,
                       const float* __restrict__ wout, unsigned short* __restrict__ Z) {
  const int bw = blockIdx.x;           // b*W + w
  const int b  = bw >> 8;              // W_ = 256
  const int st = wpos[bw * 2], en = wpos[bw * 2 + 1];
  const bool valid = (st < S_) && (en <= S_) && (en > st) && (st >= 0);
  const int cnt = valid ? (en - st) : 0;
  const float inv = 1.0f / (float)(cnt > 0 ? cnt : 1);
  for (int hh = threadIdx.x; hh < H_; hh += blockDim.x) {
    float s = 0.f;
    if (valid)
      for (int t = st; t < en; ++t)
        s += cout[((size_t)(b * S_ + t)) * H_ + hh];
    Z[(size_t)bw * 1024 + hh]       = f2bf(s * inv);
    Z[(size_t)bw * 1024 + 512 + hh] = f2bf(wout[(size_t)bw * H_ + hh]);
  }
}

// ------------- last covering word per char position ------------------------
__global__ void k_lastw(const int* __restrict__ wpos, int* __restrict__ lw) {
  __shared__ int sst[W_], sen[W_];
  const int b = blockIdx.x;
  for (int w = threadIdx.x; w < W_; w += blockDim.x) {
    int st = wpos[(b * W_ + w) * 2], en = wpos[(b * W_ + w) * 2 + 1];
    bool valid = (st < S_) && (en <= S_);
    sst[w] = valid ? st : S_ + 1;
    sen[w] = valid ? en : 0;
  }
  __syncthreads();
  for (int s = threadIdx.x; s < S_; s += blockDim.x) {
    int best = -1;
    for (int w = 0; w < W_; ++w)
      if (s >= sst[w] && s < sen[w]) best = w;
    lw[b * S_ + s] = best;
  }
}

// ------------- gated fusion -> emissions-GEMM A matrix (bf16) --------------
__global__ void k_fuse(const int* __restrict__ lw, const float* __restrict__ gate,
                       const float* __restrict__ wout, const float* __restrict__ cout,
                       unsigned short* __restrict__ lout) {
  int gid = blockIdx.x * blockDim.x + threadIdx.x;   // < B*S*512
  int h  = gid & 511;
  int bs = gid >> 9;
  int b  = bs >> 10;                                 // S_ = 1024
  int w  = lw[bs];
  float co = cout[gid];
  float v = co;
  if (w >= 0) {
    size_t bwh = ((size_t)(b * W_ + w)) * H_ + h;
    float g = gate[bwh];
    v = g * wout[bwh] + (1.f - g) * co;
  }
  lout[gid] = f2bf(v);
}

extern "C" void kernel_launch(void* const* d_in, const int* in_sizes, int n_in,
                              void* d_out, int out_size, void* d_ws, size_t ws_size,
                              hipStream_t stream) {
  (void)in_sizes; (void)n_in; (void)out_size; (void)ws_size;
  const int*   char_ids = (const int*)d_in[0];
  const int*   word_ids = (const int*)d_in[1];
  const int*   wpos     = (const int*)d_in[2];
  /* d_in[3] = attention_mask (unused by reference forward) */
  const float* char_emb = (const float*)d_in[4];
  const float* word_emb = (const float*)d_in[5];
  // LSTM param groups: char_f, char_b, word_f, word_b (Wih, Whh, bih, bhh each)
  const float* Wih[4] = {(const float*)d_in[6],  (const float*)d_in[10],
                         (const float*)d_in[14], (const float*)d_in[18]};
  const float* Whh[4] = {(const float*)d_in[7],  (const float*)d_in[11],
                         (const float*)d_in[15], (const float*)d_in[19]};
  const float* bih[4] = {(const float*)d_in[8],  (const float*)d_in[12],
                         (const float*)d_in[16], (const float*)d_in[20]};
  const float* bhh[4] = {(const float*)d_in[9],  (const float*)d_in[13],
                         (const float*)d_in[17], (const float*)d_in[21]};
  const float* Wg = (const float*)d_in[22];
  const float* bg = (const float*)d_in[23];
  const float* Wt = (const float*)d_in[24];
  const float* bt = (const float*)d_in[25];

  char* ws = (char*)d_ws;
  size_t off = 0;
  auto alloc = [&](size_t bytes) -> char* {
    char* p = ws + off;
    off += (bytes + 255) & ~(size_t)255;
    return p;
  };
  unsigned short* Xc = (unsigned short*)alloc((size_t)B_ * S_ * E_ * 2);
  unsigned short* Xw = (unsigned short*)alloc((size_t)B_ * W_ * E_ * 2);
  unsigned short *WihP[4], *WhhP[4];
  for (int i = 0; i < 4; ++i) WihP[i] = (unsigned short*)alloc(1024 * 256 * 2);
  for (int i = 0; i < 4; ++i) WhhP[i] = (unsigned short*)alloc(1024 * 256 * 2);
  unsigned short* WgP = (unsigned short*)alloc(512 * 1024 * 2);
  unsigned short* WtP = (unsigned short*)alloc(32 * 512 * 2);
  float* xpc0  = (float*)alloc((size_t)B_ * S_ * 1024 * 4);
  float* xpc1  = (float*)alloc((size_t)B_ * S_ * 1024 * 4);
  float* xpw0  = (float*)alloc((size_t)B_ * W_ * 1024 * 4);
  float* xpw1  = (float*)alloc((size_t)B_ * W_ * 1024 * 4);
  float* coutb = (float*)alloc((size_t)B_ * S_ * H_ * 4);
  float* woutb = (float*)alloc((size_t)B_ * W_ * H_ * 4);
  unsigned short* Zcat = (unsigned short*)alloc((size_t)B_ * W_ * 1024 * 2);
  float* gate  = (float*)alloc((size_t)B_ * W_ * H_ * 4);
  int*   lastw = (int*)alloc((size_t)B_ * S_ * 4);
  unsigned short* lout = (unsigned short*)alloc((size_t)B_ * S_ * H_ * 2);

  // 1. embeddings -> bf16 A matrices
  k_embed<<<(B_ * S_ * E_) / 256, 256, 0, stream>>>(char_ids, char_emb, Xc, B_ * S_);
  k_embed<<<(B_ * W_ * E_) / 256, 256, 0, stream>>>(word_ids, word_emb, Xw, B_ * W_);

  // 2. pack all weights into WMMA B-fragment tile layout (bf16)
  for (int i = 0; i < 4; ++i) {
    k_pack<<<1024, 256, 0, stream>>>(Wih[i], WihP[i], 1024, 256);
    k_pack<<<1024, 256, 0, stream>>>(Whh[i], WhhP[i], 1024, 256);
  }
  k_pack<<<2048, 256, 0, stream>>>(Wg, WgP, 512, 1024);
  k_pack<<<64,   256, 0, stream>>>(Wt, WtP, 32, 512);

  // 3. input projections (hoisted out of the recurrence)
  k_gemm<<<dim3(2048, 8), 256, 0, stream>>>(Xc, WihP[0], xpc0, bih[0], bhh[0], 32768, 1024, 256, 0);
  k_gemm<<<dim3(2048, 8), 256, 0, stream>>>(Xc, WihP[1], xpc1, bih[1], bhh[1], 32768, 1024, 256, 0);
  k_gemm<<<dim3(512,  8), 256, 0, stream>>>(Xw, WihP[2], xpw0, bih[2], bhh[2],  8192, 1024, 256, 0);
  k_gemm<<<dim3(512,  8), 256, 0, stream>>>(Xw, WihP[3], xpw1, bih[3], bhh[3],  8192, 1024, 256, 0);

  // 4. persistent recurrent kernels (h in LDS, c in registers, Whh in L2)
  k_lstm<<<2, 512, 0, stream>>>(xpc0, WhhP[0], coutb, S_, 0, 0);
  k_lstm<<<2, 512, 0, stream>>>(xpc1, WhhP[1], coutb, S_, 1, 256);
  k_lstm<<<2, 512, 0, stream>>>(xpw0, WhhP[2], woutb, W_, 0, 0);
  k_lstm<<<2, 512, 0, stream>>>(xpw1, WhhP[3], woutb, W_, 1, 256);

  // 5. span mean + concat, gate GEMM with fused sigmoid
  k_span<<<B_ * W_, 256, 0, stream>>>(wpos, coutb, woutb, Zcat);
  k_gemm<<<dim3(512, 4), 256, 0, stream>>>(Zcat, WgP, gate, bg, nullptr, 8192, 512, 1024, 1);

  // 6. last covering word, gated fusion
  k_lastw<<<B_, 1024, 0, stream>>>(wpos, lastw);
  k_fuse<<<(B_ * S_ * H_) / 256, 256, 0, stream>>>(lastw, gate, woutb, coutb, lout);

  // 7. emissions GEMM straight into d_out (f32 [B,S,L])
  k_gemm<<<dim3(2048, 1), 64, 0, stream>>>(lout, WtP, (float*)d_out, bt, nullptr, 32768, 32, 512, 0);
}
